// NeuralSurface_13993003450555
// MI455X (gfx1250) — matrix operand order repaired and checked by reading, not compiled, not run on hardware
//
#include <hip/hip_runtime.h>

// CDNA5 / gfx1250 fully-fused NeRF-SDF MLP using v_wmma_f32_16x16x32_f16.
// v2: 4x M-blocking per wave (64 rows) so every 1KB B-fragment from L2 feeds
// 4 WMMAs -> ~4x less L2 weight traffic (16GB -> 4GB per pass, ~compute
// balanced). A-fragments are re-read from a conflict-free LDS slab per use;
// hidden activations ping-pong between two 256-wide LDS buffers since output
// can no longer overwrite input in-place. B-fragments for nt+2 are prefetched
// (global_prefetch_b8) to hide L2 latency at low wave occupancy.

typedef __attribute__((ext_vector_type(16))) _Float16 v16h;
typedef __attribute__((ext_vector_type(8)))  _Float16 v8h;
typedef __attribute__((ext_vector_type(8)))  float    v8f;

#define HID 256
#define NPTS 262144
// Slab: 64 (padded emb) | 256 (hidden buf A) | 256 (hidden buf B) | 8 pad.
// 584 halfs = 292 DW; 292 % 64 = 36, gcd(36,64)=4 -> the 16-row ds_load_b128
// pattern hits all 64 LDS banks exactly once (conflict-free).
#define SLAB_W 584
#define BUF0 64
#define BUF1 320
#define MT 4                     // M-tiles (16 rows each) per wave
#define ROWS_PER_WAVE (MT * 16)  // 64
#define WAVES_PER_BLOCK 2
#define ROWS_PER_BLOCK (WAVES_PER_BLOCK * ROWS_PER_WAVE)  // 128

// Fragment-layout weight offsets in halfs (16 N-tiles * KT * 512 per layer).
// KT per layer: {2,8,8,8,10,8,8,8}  (K padded: 64, 256x3, 320, 256x3)
__host__ __device__ constexpr int layer_off(int l) {
  constexpr int off[8] = {0, 16384, 81920, 147456, 212992, 294912, 360448, 425984};
  return off[l];
}

struct Params {
  const float* w[8];
  const float* b[8];
};

// ---------------------------------------------------------------------------
// Prep: f32 [d_in, 256] row-major weights -> f16 B-fragment layout.
// Fragment (kTile,nTile) = 32 lanes x 16 halfs contiguous per lane:
//   lane L: col N = nTile*16 + (L&15), K = kTile*32 + (L>>4)*8 + (j<8 ? j : j+8), j=0..15
// K-padding / concat remap handled here so the main kernel K-loop is clean.
// ---------------------------------------------------------------------------
__global__ void prep_weights(Params p, _Float16* __restrict__ wf) {
  const int l  = blockIdx.y;
  const int KT = (l == 0) ? 2 : ((l == 4) ? 10 : 8);
  const int total = 16 * KT * 512;
  const int idx = blockIdx.x * 256 + threadIdx.x;
  if (idx >= total) return;

  const int nt   = idx / (KT * 512);
  const int rem  = idx - nt * (KT * 512);
  const int kt   = rem >> 9;
  const int lane = (rem >> 4) & 31;
  const int j    = rem & 15;
  const int N    = nt * 16 + (lane & 15);
  const int k    = kt * 32 + (lane >> 4) * 8 + (j < 8 ? j : j + 8);

  int r;
  if (l == 0)      r = (k < 39) ? k : -1;                          // pad 39->64
  else if (l == 4) r = (k < 39) ? k : ((k >= 64) ? (k - 25) : -1); // concat pad
  else             r = k;                                          // dense 256

  const float v = (r >= 0) ? p.w[l][r * HID + N] : 0.0f;
  wf[layer_off(l) + idx] = (_Float16)v;
}

// ---------------------------------------------------------------------------
// One dense layer for one wave's 64-row slab (4 M-tiles):
//   X[64 x 32*KT] (LDS f16) @ W[32*KT x 256] (global frags) + b -> ReLU -> LDS
// Each B-fragment load feeds MT=4 WMMAs. First EKT k-tiles read from the
// embedding region (col 0), the rest from the live hidden buffer at inOff.
// ---------------------------------------------------------------------------
template <int KT, int EKT>
__device__ __forceinline__ void mlp_layer(_Float16* __restrict__ slab,
                                          int inOff, int outOff,
                                          const _Float16* __restrict__ wfrag,
                                          const float* __restrict__ bias, int lane) {
  const int rowA   = lane & 15;
  const int kSel   = (lane >> 4) << 3;  // A layout: lanes<16: K0-7/16-23; else K8-15/24-31
  const int colT   = lane & 15;
  const int rowSel = (lane >> 4) << 3;  // D layout: lanes>=16 hold M = r+8

  for (int nt = 0; nt < 16; ++nt) {
    v8f acc[MT] = {};
    const _Float16* wb = wfrag + nt * (KT * 512) + lane * 16;
#pragma unroll
    for (int kt = 0; kt < KT; ++kt) {
      v16h b = *(const v16h*)(wb + kt * 512);      // 32B/lane, coalesced 1KB frag
      __builtin_prefetch(wb + kt * 512 + 2 * KT * 512, 0, 1);  // nt+2 frags
      const int kcol = (kt < EKT) ? (kt * 32) : (inOff + (kt - EKT) * 32);
#pragma unroll
      for (int m = 0; m < MT; ++m) {
        const _Float16* abase = slab + (m * 16 + rowA) * SLAB_W + kcol + kSel;
        v8h lo = *(const v8h*)(abase);             // ds_load_b128, conflict-free
        v8h hi = *(const v8h*)(abase + 16);        // ds_load_b128
        v16h a = __builtin_shufflevector(lo, hi, 0, 1, 2, 3, 4, 5, 6, 7,
                                                 8, 9, 10, 11, 12, 13, 14, 15);
        acc[m] = __builtin_amdgcn_wmma_f32_16x16x32_f16(
                     /*neg_a=*/false, a, /*neg_b=*/false, b,
                     /*c_mod=*/(short)0, acc[m], /*reuse=*/false, false);
      }
    }
    const float bv = bias[nt * 16 + colT];
#pragma unroll
    for (int m = 0; m < MT; ++m) {
      _Float16* op = slab + outOff + nt * 16 + colT + (m * 16 + rowSel) * SLAB_W;
#pragma unroll
      for (int r = 0; r < 8; ++r) {
        float v = acc[m][r] + bv;
        v = v > 0.0f ? v : 0.0f;                   // fused ReLU
        op[r * SLAB_W] = (_Float16)v;              // packs to ds_store_b16(_d16_hi)
      }
    }
  }
}

// ---------------------------------------------------------------------------
// Fused forward pass: embed -> 8 layers (skip-concat before layer 4) -> sdf.
// Per-wave-private slab => no barriers (LDS ops are in-order within a wave).
// ---------------------------------------------------------------------------
__global__ __launch_bounds__(WAVES_PER_BLOCK * 32)
void NeuralSurface_13993003450555_kernel(const float* __restrict__ points,
                                         const _Float16* __restrict__ wf,
                                         Params p,
                                         const float* __restrict__ wsdf,
                                         const float* __restrict__ bsdf,
                                         float* __restrict__ out) {
  __shared__ alignas(16) _Float16 slab_s[WAVES_PER_BLOCK][ROWS_PER_WAVE * SLAB_W];
  const int lane = threadIdx.x & 31;
  const int wv   = threadIdx.x >> 5;
  _Float16* slab = slab_s[wv];
  const int rowBase = blockIdx.x * ROWS_PER_BLOCK + wv * ROWS_PER_WAVE;

  // Harmonic embedding: cols [0..17]=sin, [18..35]=cos, [36..38]=xyz, 39..63=0.
  {
    const int row16 = lane & 15;
    const int half  = lane >> 4;  // two lanes cooperate per row
#pragma unroll
    for (int m = 0; m < MT; ++m) {
      const int row = m * 16 + row16;
      const float* pp = points + (size_t)(rowBase + row) * 3;
      const float p0 = pp[0], p1 = pp[1], p2 = pp[2];
      _Float16* er = slab + row * SLAB_W;
      for (int c = 39 + half; c < 64; c += 2) er[c] = (_Float16)0.0f;
      for (int t = 0; t < 20; ++t) {
        const int col = half * 20 + t;
        if (col >= 39) break;
        float v;
        if (col < 18) {
          const int i = col / 6;
          const float x = (i == 0) ? p0 : ((i == 1) ? p1 : p2);
          v = __sinf(x * (float)(1 << (col % 6)));
        } else if (col < 36) {
          const int cc = col - 18;
          const int i = cc / 6;
          const float x = (i == 0) ? p0 : ((i == 1) ? p1 : p2);
          v = __cosf(x * (float)(1 << (cc % 6)));
        } else {
          v = (col == 36) ? p0 : ((col == 37) ? p1 : p2);
        }
        er[col] = (_Float16)v;
      }
    }
  }

  // Ping-pong hidden buffers: L0 -> BUF0, then alternate each layer.
  mlp_layer<2, 2>(slab, 0,    BUF0, wf + layer_off(0), p.b[0], lane); // emb -> 256
  mlp_layer<8, 0>(slab, BUF0, BUF1, wf + layer_off(1), p.b[1], lane);
  mlp_layer<8, 0>(slab, BUF1, BUF0, wf + layer_off(2), p.b[2], lane);
  mlp_layer<8, 0>(slab, BUF0, BUF1, wf + layer_off(3), p.b[3], lane);
  mlp_layer<10, 2>(slab, BUF1, BUF0, wf + layer_off(4), p.b[4], lane); // concat 295->320
  mlp_layer<8, 0>(slab, BUF0, BUF1, wf + layer_off(5), p.b[5], lane);
  mlp_layer<8, 0>(slab, BUF1, BUF0, wf + layer_off(6), p.b[6], lane);
  mlp_layer<8, 0>(slab, BUF0, BUF1, wf + layer_off(7), p.b[7], lane);

  // Final 256 -> 1: scalar dot from BUF1 (WMMA would waste 15/16 of the tile).
  {
    const int row16 = lane & 15;
    const int half  = lane >> 4;
    const float* wv2 = wsdf + half * 128;
#pragma unroll
    for (int m = 0; m < MT; ++m) {
      const int row = m * 16 + row16;
      const _Float16* hr = slab + row * SLAB_W + BUF1 + half * 128;
      float s = 0.0f;
#pragma unroll 8
      for (int j2 = 0; j2 < 128; ++j2) s += (float)hr[j2] * wv2[j2];
      s += __shfl_xor(s, 16, 32);  // combine the two half-lane partial sums
      if (half == 0) out[rowBase + row] = s + bsdf[0];
    }
  }
}

extern "C" void kernel_launch(void* const* d_in, const int* in_sizes, int n_in,
                              void* d_out, int out_size, void* d_ws, size_t ws_size,
                              hipStream_t stream) {
  (void)in_sizes; (void)n_in; (void)out_size; (void)ws_size;
  const float* points = (const float*)d_in[0];
  Params p;
  for (int i = 0; i < 8; ++i) {
    p.w[i] = (const float*)d_in[1 + 2 * i];
    p.b[i] = (const float*)d_in[2 + 2 * i];
  }
  const float* wsdf = (const float*)d_in[17];
  const float* bsdf = (const float*)d_in[18];
  float* out = (float*)d_out;
  _Float16* wf = (_Float16*)d_ws;  // 491520 halfs = 960 KB of scratch

  // Swizzle all weights into WMMA B-fragment layout (f16).
  prep_weights<<<dim3(320, 8), 256, 0, stream>>>(p, wf);

  // Fused MLP: 2048 blocks x 2 waves x 64 rows = 262144 points.
  NeuralSurface_13993003450555_kernel<<<dim3(NPTS / ROWS_PER_BLOCK),
                                        WAVES_PER_BLOCK * 32, 0, stream>>>(
      points, wf, p, wsdf, bsdf, out);
}
